// MoRALayer_8143257994015
// MI455X (gfx1250) — matrix operand order repaired and checked by reading, not compiled, not run on hardware
//
#include <hip/hip_runtime.h>

// Problem dims (fixed by setup_inputs): d = 4096, mora m = 2048, tokens = 4*2048 = 8192
#define D   4096
#define MD  2048
#define BM  128
#define BN  128
#define BK  32
#define LDT 40   // BK + 8 halves pad -> 80B row stride, conflict-free b128 frag reads
                 // (TDM pad: 4 DWORDs after every 16 DWORDs reproduces this layout)

typedef __attribute__((ext_vector_type(16))) __bf16 v16bf;
typedef __attribute__((ext_vector_type(8)))  __bf16 v8bf;
typedef __attribute__((ext_vector_type(4)))  __bf16 v4bf;
typedef __attribute__((ext_vector_type(8)))  float  v8f;
typedef __attribute__((ext_vector_type(4)))  float  f32x4;
typedef unsigned int u32x4 __attribute__((ext_vector_type(4)));
typedef int          i32x4 __attribute__((ext_vector_type(4)));
typedef int          i32x8 __attribute__((ext_vector_type(8)));

union Frag16 { v16bf v; v8bf h[2]; };

// fp32 -> bf16 hi/lo split: f ~= hi + lo with |err| ~ 2^-16 |f|
__device__ __forceinline__ void split_bf16(float f, __bf16* hi, __bf16* lo) {
  unsigned u  = __builtin_bit_cast(unsigned, f);
  float    hf = __builtin_bit_cast(float, u & 0xFFFF0000u);
  float    r  = f - hf;
  unsigned ru = __builtin_bit_cast(unsigned, r);
  *hi = __builtin_bit_cast(__bf16, (unsigned short)(u >> 16));
  *lo = __builtin_bit_cast(__bf16, (unsigned short)(ru >> 16));
}

// ---------------- prep kernels ----------------

__global__ void zero_f(float* __restrict__ p, int n) {
  int t = blockIdx.x * blockDim.x + threadIdx.x;
  if (t < n) p[t] = 0.0f;
}

// column sum-of-squares of (weight + tile(mora)) via atomics; no store
__global__ void mora_colnorm(const float* __restrict__ weight,
                             const float* __restrict__ mora,
                             float* __restrict__ sums) {
  const int i  = blockIdx.x * 256 + threadIdx.x;  // column (d_in)
  const int o0 = blockIdx.y * 256;                // row chunk (d_out)
  float sum = 0.0f;
  #pragma unroll 4
  for (int o = o0; o < o0 + 256; ++o) {
    float w = weight[(size_t)o * D + i] + mora[(size_t)(o >> 1) * MD + (i >> 1)];
    sum = fmaf(w, w, sum);
  }
  atomicAdd(&sums[i], sum);
}

__global__ void mora_colscale(const float* __restrict__ dmag,
                              const float* __restrict__ sums,
                              float* __restrict__ sc) {
  int i = blockIdx.x * 256 + threadIdx.x;
  sc[i] = dmag[i] / sqrtf(sums[i]);
}

// wh/wl = split((weight + tile(mora)) * sc[col]), one float4 per thread
__global__ void mora_wsplit(const float* __restrict__ weight,
                            const float* __restrict__ mora,
                            const float* __restrict__ sc,
                            __bf16* __restrict__ wh, __bf16* __restrict__ wl) {
  const size_t base = ((size_t)blockIdx.x * 256 + threadIdx.x) * 4;
  const int o  = (int)(base >> 12);        // D == 4096
  const int i0 = (int)(base & (size_t)(D - 1));
  f32x4 wv = *(const f32x4*)(weight + base);
  f32x4 sv = *(const f32x4*)(sc + i0);
  const float* mp = mora + (size_t)(o >> 1) * MD + (i0 >> 1);
  const float m0v = mp[0], m1v = mp[1];
  v4bf h4, l4;
  #pragma unroll
  for (int e = 0; e < 4; ++e) {
    float w = (wv[e] + ((e < 2) ? m0v : m1v)) * sv[e];
    __bf16 h, l;
    split_bf16(w, &h, &l);
    h4[e] = h; l4[e] = l;
  }
  *(v4bf*)(wh + base) = h4;
  *(v4bf*)(wl + base) = l4;
}

// xh/xl = split(x), one float4 per thread
__global__ void mora_xsplit(const float* __restrict__ x,
                            __bf16* __restrict__ xh, __bf16* __restrict__ xl) {
  const size_t base = ((size_t)blockIdx.x * 256 + threadIdx.x) * 4;
  f32x4 v = *(const f32x4*)(x + base);
  v4bf h4, l4;
  #pragma unroll
  for (int e = 0; e < 4; ++e) {
    __bf16 h, l;
    split_bf16(v[e], &h, &l);
    h4[e] = h; l4[e] = l;
  }
  *(v4bf*)(xh + base) = h4;
  *(v4bf*)(xl + base) = l4;
}

// ---------------- TDM tile load: 128 rows x 32 bf16, row stride D, LDS pad 4dw/16dw ----

__device__ __forceinline__ void tdm_tile_load(unsigned long long ga, unsigned lds_off,
                                              unsigned rows_total) {
  // D# group 0: count=1 | lds_addr | global_addr[56:0] | type=2
  u32x4 g0 = { 1u, lds_off, (unsigned)ga, (unsigned)(ga >> 32) | (2u << 30) };
  // D# group 1: data_size=2B(code 1), pad_enable, pad_interval=16dw(code 3),
  // pad_amount=4dw(code 3); tensor_dim0=D, tensor_dim1=rows_total;
  // tile_dim0=BK, tile_dim1=BM; tensor_dim0_stride=D
  i32x8 g1 = {
    (int)((1u << 16) | (1u << 20) | (3u << 22) | (3u << 25)),
    (int)(((unsigned)D & 0xFFFFu) << 16),
    (int)((((unsigned)D >> 16) & 0xFFFFu) | ((rows_total & 0xFFFFu) << 16)),
    (int)(((rows_total >> 16) & 0xFFFFu) | ((unsigned)BK << 16)),
    (int)BM,
    (int)D,
    0, 0 };
  i32x4 z4 = { 0, 0, 0, 0 };
#if defined(__clang_major__) && (__clang_major__ >= 23)
  i32x8 z8 = { 0, 0, 0, 0, 0, 0, 0, 0 };
  __builtin_amdgcn_tensor_load_to_lds(g0, g1, z4, z4, z8, 0);
#else
  __builtin_amdgcn_tensor_load_to_lds(g0, g1, z4, z4, 0);
#endif
}

// ---------------- main GEMM: out = X(8192xK) * W(NxK)^T, bf16x3 split on WMMA --------

__launch_bounds__(256)
__global__ void mora_gemm(const __bf16* __restrict__ Xh, const __bf16* __restrict__ Xl,
                          const __bf16* __restrict__ Wh, const __bf16* __restrict__ Wl,
                          float* __restrict__ out, int Mrows) {
  extern __shared__ __align__(16) unsigned char smem_raw[];
  __bf16* lds = (__bf16*)smem_raw;
  constexpr int TILE = BM * LDT;  // 5120 halves = 10240 B per plane per buffer
  // plane order (x2 buffers each): Ah(0) Al(1) Bh(2) Bl(3)

  const int tid  = threadIdx.x;
  const int lane = tid & 31;
  const int wave = tid >> 5;
  const int wm   = wave & 3;          // 4 waves over M -> 32-row strip each
  const int wn   = wave >> 2;         // 2 waves over N -> 64-col strip each
  const int lr   = lane & 15;
  const int hi16 = lane >> 4;

  const size_t m0 = (size_t)blockIdx.y * BM;
  const size_t n0 = (size_t)blockIdx.x * BN;

  const unsigned long long gAh = (unsigned long long)(uintptr_t)Xh + m0 * D * 2ull;
  const unsigned long long gAl = (unsigned long long)(uintptr_t)Xl + m0 * D * 2ull;
  const unsigned long long gBh = (unsigned long long)(uintptr_t)Wh + n0 * D * 2ull;
  const unsigned long long gBl = (unsigned long long)(uintptr_t)Wl + n0 * D * 2ull;

  auto issue = [&](int buf, int kt) {
    const unsigned long long ko = (unsigned long long)kt * BK * 2ull;  // byte offset along k
    tdm_tile_load(gAh + ko, (unsigned)((0 * 2 + buf) * TILE * 2), (unsigned)Mrows);
    tdm_tile_load(gAl + ko, (unsigned)((1 * 2 + buf) * TILE * 2), (unsigned)Mrows);
    tdm_tile_load(gBh + ko, (unsigned)((2 * 2 + buf) * TILE * 2), (unsigned)D);
    tdm_tile_load(gBl + ko, (unsigned)((3 * 2 + buf) * TILE * 2), (unsigned)D);
  };

  v8f acc[2][4] = {};

  if (wave == 0) {
    issue(0, 0);
    __builtin_amdgcn_s_wait_tensorcnt(0);
  }
  __syncthreads();

  const int NK = D / BK;  // 128
  for (int kt = 0; kt < NK; ++kt) {
    const int cur = kt & 1;
    if (wave == 0 && kt + 1 < NK) issue((kt + 1) & 1, kt + 1);

    const __bf16* pAh = lds + (0 * 2 + cur) * TILE;
    const __bf16* pAl = lds + (1 * 2 + cur) * TILE;
    const __bf16* pBh = lds + (2 * 2 + cur) * TILE;
    const __bf16* pBl = lds + (3 * 2 + cur) * TILE;

    // A fragments: lane row = base + (lane&15); lanes>=16 hold K=8..15,24..31
    Frag16 afh[2], afl[2];
    const int ka = hi16 ? 8 : 0;
    #pragma unroll
    for (int mt = 0; mt < 2; ++mt) {
      const int rowoff = (wm * 32 + mt * 16 + lr) * LDT;
      afh[mt].h[0] = *(const v8bf*)(pAh + rowoff + ka);
      afh[mt].h[1] = *(const v8bf*)(pAh + rowoff + ka + 16);
      afl[mt].h[0] = *(const v8bf*)(pAl + rowoff + ka);
      afl[mt].h[1] = *(const v8bf*)(pAl + rowoff + ka + 16);
    }

    // B fragments: lane col = base + (lane&15); lanes>=16 hold K=8..15,24..31
    const int kb = hi16 ? 8 : 0;
    #pragma unroll
    for (int nt = 0; nt < 4; ++nt) {
      Frag16 bfh, bfl;
      const int coloff = (wn * 64 + nt * 16 + lr) * LDT;
      bfh.h[0] = *(const v8bf*)(pBh + coloff + kb);
      bfh.h[1] = *(const v8bf*)(pBh + coloff + kb + 16);
      bfl.h[0] = *(const v8bf*)(pBl + coloff + kb);
      bfl.h[1] = *(const v8bf*)(pBl + coloff + kb + 16);
      #pragma unroll
      for (int mt = 0; mt < 2; ++mt) {
        // small terms first, dominant last (all f32 accumulate)
        acc[mt][nt] = __builtin_amdgcn_wmma_f32_16x16x32_bf16(
            false, afl[mt].v, false, bfh.v, (short)0, acc[mt][nt], false, false);
        acc[mt][nt] = __builtin_amdgcn_wmma_f32_16x16x32_bf16(
            false, afh[mt].v, false, bfl.v, (short)0, acc[mt][nt], false, false);
        acc[mt][nt] = __builtin_amdgcn_wmma_f32_16x16x32_bf16(
            false, afh[mt].v, false, bfh.v, (short)0, acc[mt][nt], false, false);
      }
    }

    if (wave == 0 && kt + 1 < NK) __builtin_amdgcn_s_wait_tensorcnt(0);
    __syncthreads();
  }

  // C/D layout: VGPR r -> M = r + 8*hi16; lane%16 -> N
  #pragma unroll
  for (int mt = 0; mt < 2; ++mt) {
    #pragma unroll
    for (int nt = 0; nt < 4; ++nt) {
      const size_t row0 = m0 + (size_t)(wm * 32 + mt * 16 + hi16 * 8);
      const size_t col  = n0 + (size_t)(wn * 64 + nt * 16 + lr);
      #pragma unroll
      for (int rr = 0; rr < 8; ++rr)
        out[(row0 + rr) * D + col] = acc[mt][nt][rr];
    }
  }
}

// ---------------- launcher ----------------

extern "C" void kernel_launch(void* const* d_in, const int* in_sizes, int n_in,
                              void* d_out, int out_size, void* d_ws, size_t ws_size,
                              hipStream_t stream) {
  const float* x      = (const float*)d_in[0];
  const float* weight = (const float*)d_in[1];
  const float* mora   = (const float*)d_in[2];
  const float* dmag   = (const float*)d_in[3];
  float* out = (float*)d_out;

  const int    Mrows  = in_sizes[0] / D;        // 8192
  const size_t xelems = (size_t)Mrows * D;
  const size_t welems = (size_t)D * D;

  // workspace: xh | xl | wh | wl (bf16 planes) | sums | sc  (~192 MB)
  __bf16* xh = (__bf16*)d_ws;
  __bf16* xl = xh + xelems;
  __bf16* wh = xl + xelems;
  __bf16* wl = wh + welems;
  float* sums = (float*)(wl + welems);
  float* sc   = sums + D;

  zero_f<<<(D + 255) / 256, 256, 0, stream>>>(sums, D);
  mora_colnorm<<<dim3(D / 256, D / 256), 256, 0, stream>>>(weight, mora, sums);
  mora_colscale<<<D / 256, 256, 0, stream>>>(dmag, sums, sc);
  mora_wsplit<<<(int)(welems / 4 / 256), 256, 0, stream>>>(weight, mora, sc, wh, wl);
  mora_xsplit<<<(int)(xelems / 4 / 256), 256, 0, stream>>>(x, xh, xl);

  constexpr size_t SMEM = (size_t)4 * 2 * BM * LDT * sizeof(__bf16);  // 81920 B
  dim3 grid(D / BN, Mrows / BM);
  mora_gemm<<<grid, 256, SMEM, stream>>>(xh, xl, wh, wl, out, Mrows);
}
